// QuantumTransformer_65481071395841
// MI455X (gfx1250) — matrix operand order repaired and checked by reading, not compiled, not run on hardware
//
#include <hip/hip_runtime.h>
#include <math.h>

// ---------------- problem constants ----------------
static constexpr int Bc = 4, Sc = 1024, Dc = 1024, Hc = 16, Lc = 8, Fc = 4096;
static constexpr int DKc = Dc / Hc; // 64
static constexpr int BS = Bc * Sc;  // 4096 rows

typedef __bf16 bf16_t;
typedef __bf16 v16bf __attribute__((ext_vector_type(16)));
typedef float  v8f   __attribute__((ext_vector_type(8)));
typedef int    v4i   __attribute__((ext_vector_type(4)));
typedef __attribute__((address_space(1))) v4i* as1_v4i_p;
typedef __attribute__((address_space(3))) v4i* as3_v4i_p;

union Frag16 { v16bf v; unsigned short s[16]; unsigned int u[8]; };

__device__ __forceinline__ float sigmf(float x) { return 1.f / (1.f + expf(-x)); }

#if __has_builtin(__builtin_amdgcn_global_load_async_to_lds_b128) && \
    __has_builtin(__builtin_amdgcn_s_wait_asynccnt)
#define USE_ASYNC_LDS 1
#else
#define USE_ASYNC_LDS 0
#endif

// ---------------- embedding + hybrid positional encoding ----------------
__global__ void k_embed(const int* __restrict__ tok, const float* __restrict__ emb,
                        const float* __restrict__ pe_qw, float* __restrict__ x) {
  const int bs = blockIdx.x;          // 0..B*S-1
  const int s  = bs % Sc;
  const int t  = tok[bs];
  const float wpe = sigmf(pe_qw[0]);
  const float c0 = -logf(10000.f) / (float)Dc;
  for (int d = threadIdx.x; d < Dc; d += blockDim.x) {
    const int p = d >> 1;
    const float dv = expf((float)(2 * p) * c0);
    const float ang = (float)s * dv;
    const float pe  = (d & 1) ? cosf(ang) : sinf(ang);
    const float o   = emb[(long long)t * Dc + d] + pe;
    x[(long long)bs * Dc + d] = wpe * sinf(o) + (1.f - wpe) * o;
  }
}

// ---------------- fp32 -> bf16 convert, 4 elements/thread ----------------
__global__ void k_cvt4(const float* __restrict__ in, bf16_t* __restrict__ out, int n4) {
  const int i = blockIdx.x * blockDim.x + threadIdx.x;
  if (i < n4) {
    const float4 f = ((const float4*)in)[i];
    bf16_t o[4] = {(bf16_t)f.x, (bf16_t)f.y, (bf16_t)f.z, (bf16_t)f.w};
    *(uint2*)(out + 4ll * i) = *(uint2*)o;
  }
}

// ---------------- fp32 [K,N] -> bf16 transposed [N,K] (LDS-tiled) ----------------
__global__ __launch_bounds__(256)
void k_cvtT(const float* __restrict__ in, bf16_t* __restrict__ out, int K, int N) {
  __shared__ float t[32][33];
  const int tid = threadIdx.x;
  const int k0 = blockIdx.x * 32, n0 = blockIdx.y * 32;
  for (int i = tid; i < 1024; i += 256) {
    const int r = i >> 5, c = i & 31;
    t[r][c] = in[(long long)(k0 + r) * N + n0 + c];
  }
  __syncthreads();
  for (int i = tid; i < 1024; i += 256) {
    const int r = i >> 5, c = i & 31;
    out[(long long)(n0 + r) * K + k0 + c] = (bf16_t)t[c][r];
  }
}

// ---------------- bf16 WMMA GEMM: C[M,N] = A[M,K] * BT[N,K]^T ----------------
// grid = (N/64, M/128), block = 256 (8 waves; each wave: one 32x32 macro-tile)
// Double-buffered LDS with async-to-LDS loads overlapping WMMA.
// mode 0: f32 (+bias)   mode 1: bf16
// mode 2: bf16 of wf*sin(t)+(1-wf)*relu(t), t=c+bias   (hybrid FFN act)
// mode 3: bf16 stored transposed per-head: VT[((b*H+h)*DK+dk)*S + s]
__global__ __launch_bounds__(256)
void k_gemm(const bf16_t* __restrict__ A, const bf16_t* __restrict__ BT,
            void* __restrict__ Cout, const float* __restrict__ bias,
            const float* __restrict__ gate,
            int K, int lda, int ldc, int mode) {
  __shared__ __align__(16) bf16_t As[2][128][40];  // [stage][m][k]
  __shared__ __align__(16) bf16_t Bs[2][64][40];   // [stage][n][k]
  const int tid  = threadIdx.x;
  const int lane = tid & 31, wave = tid >> 5;
  const int row0 = blockIdx.y * 128, col0 = blockIdx.x * 64;
  const int tm = (wave >> 1) * 32, tn = (wave & 1) * 32;
  const int n = lane & 15, half = lane >> 4;
  const int ra = tid >> 2, ca = (tid & 3) * 8;     // b128 slots

  v8f acc[2][2];
#pragma unroll
  for (int i = 0; i < 2; ++i)
#pragma unroll
    for (int j = 0; j < 2; ++j)
      acc[i][j] = (v8f){0.f,0.f,0.f,0.f,0.f,0.f,0.f,0.f};

  auto issue = [&](int st, int k0) {
#if USE_ASYNC_LDS
    __builtin_amdgcn_global_load_async_to_lds_b128(
        (as1_v4i_p)&A[(long long)(row0 + ra) * lda + k0 + ca],
        (as3_v4i_p)&As[st][ra][ca], 0, 0);
    __builtin_amdgcn_global_load_async_to_lds_b128(
        (as1_v4i_p)&A[(long long)(row0 + 64 + ra) * lda + k0 + ca],
        (as3_v4i_p)&As[st][64 + ra][ca], 0, 0);
    __builtin_amdgcn_global_load_async_to_lds_b128(
        (as1_v4i_p)&BT[(long long)(col0 + ra) * K + k0 + ca],
        (as3_v4i_p)&Bs[st][ra][ca], 0, 0);
#else
    *(uint4*)&As[st][ra][ca]      = *(const uint4*)&A[(long long)(row0 + ra) * lda + k0 + ca];
    *(uint4*)&As[st][64 + ra][ca] = *(const uint4*)&A[(long long)(row0 + 64 + ra) * lda + k0 + ca];
    *(uint4*)&Bs[st][ra][ca]      = *(const uint4*)&BT[(long long)(col0 + ra) * K + k0 + ca];
#endif
  };

  issue(0, 0);
  int cur = 0;
  for (int k0 = 0; k0 < K; k0 += 32) {
    if (k0 + 32 < K) {
      issue(cur ^ 1, k0 + 32);   // next stage in flight during compute
#if USE_ASYNC_LDS
      __builtin_amdgcn_s_wait_asynccnt(3);   // wait only for current stage
#endif
    } else {
#if USE_ASYNC_LDS
      __builtin_amdgcn_s_wait_asynccnt(0);
#endif
    }
    __syncthreads();

    Frag16 aF[2], bF[2];
#pragma unroll
    for (int i = 0; i < 2; ++i) {
      const bf16_t* arow = &As[cur][tm + 16 * i + n][0];
      ((uint4*)aF[i].u)[0] = *(const uint4*)(arow + 8 * half);
      ((uint4*)aF[i].u)[1] = *(const uint4*)(arow + 16 + 8 * half);
      const bf16_t* brow = &Bs[cur][tn + 16 * i + n][0];
      ((uint4*)bF[i].u)[0] = *(const uint4*)(brow + 8 * half);
      ((uint4*)bF[i].u)[1] = *(const uint4*)(brow + 16 + 8 * half);
    }
#pragma unroll
    for (int i = 0; i < 2; ++i)
#pragma unroll
      for (int j = 0; j < 2; ++j)
        acc[i][j] = __builtin_amdgcn_wmma_f32_16x16x32_bf16(
            false, aF[i].v, false, bF[j].v, (short)0, acc[i][j], false, false);
    __syncthreads();   // all reads of `cur` done before it is overwritten
    cur ^= 1;
  }

#pragma unroll
  for (int i = 0; i < 2; ++i)
#pragma unroll
    for (int j = 0; j < 2; ++j)
#pragma unroll
      for (int v = 0; v < 8; ++v) { // C/D layout: M = v + 8*half, N = lane&15
        const int rr = row0 + tm + 16 * i + v + 8 * half;
        const int cc = col0 + tn + 16 * j + n;
        const float val = acc[i][j][v];
        if (mode == 0) {
          ((float*)Cout)[(long long)rr * ldc + cc] = bias ? val + bias[cc] : val;
        } else if (mode == 1) {
          ((bf16_t*)Cout)[(long long)rr * ldc + cc] = (bf16_t)val;
        } else if (mode == 2) {
          const float t  = val + bias[cc];
          const float wf = sigmf(gate[0]);
          const float h  = wf * sinf(t) + (1.f - wf) * fmaxf(t, 0.f);
          ((bf16_t*)Cout)[(long long)rr * ldc + cc] = (bf16_t)h;
        } else { // mode 3: per-head transposed V store
          const int bb = rr >> 10, s = rr & 1023;
          const int hh = cc >> 6, dk = cc & 63;
          ((bf16_t*)Cout)[((long long)(bb * Hc + hh) * DKc + dk) * Sc + s] = (bf16_t)val;
        }
      }
}

// ---------------- attention: one block per (b, head, 16-row q tile) ----------------
__global__ __launch_bounds__(256)
void k_attn(const bf16_t* __restrict__ qb, const bf16_t* __restrict__ kbuf,
            const bf16_t* __restrict__ vt, float* __restrict__ ob,
            const float* __restrict__ gates) {
  __shared__ __align__(16) bf16_t pb[16][1032];  // probabilities, bf16
  __shared__ float red[8][16];
  __shared__ float rowstat[16];
  const int tid = threadIdx.x, lane = tid & 31, wave = tid >> 5;
  const int q0 = blockIdx.x * 16;
  const int hh = blockIdx.y;
  const int bb = blockIdx.z;
  const int n = lane & 15, half = lane >> 4;
  const long long baseH = ((long long)bb * Sc) * Dc + (long long)hh * DKc;

  // A frags from q (K = DK = 64 -> two 32-wide k-steps), 128-bit loads
  Frag16 a0, a1;
  {
    const bf16_t* qrow = qb + baseH + (long long)(q0 + n) * Dc;
    ((uint4*)a0.u)[0] = *(const uint4*)(qrow + 8 * half);
    ((uint4*)a0.u)[1] = *(const uint4*)(qrow + 16 + 8 * half);
    ((uint4*)a1.u)[0] = *(const uint4*)(qrow + 32 + 8 * half);
    ((uint4*)a1.u)[1] = *(const uint4*)(qrow + 48 + 8 * half);
  }

  // ---- scores: each wave computes 8 of the 64 16x16 n-tiles ----
  v8f acc[8];
#pragma unroll
  for (int t = 0; t < 8; ++t) acc[t] = (v8f){0.f,0.f,0.f,0.f,0.f,0.f,0.f,0.f};
  for (int t = 0; t < 8; ++t) {
    const int ncol = (wave + t * 8) * 16 + n;
    const bf16_t* krow = kbuf + baseH + (long long)ncol * Dc;
    Frag16 b0, b1;
    ((uint4*)b0.u)[0] = *(const uint4*)(krow + half * 8);
    ((uint4*)b0.u)[1] = *(const uint4*)(krow + 16 + half * 8);
    ((uint4*)b1.u)[0] = *(const uint4*)(krow + 32 + half * 8);
    ((uint4*)b1.u)[1] = *(const uint4*)(krow + 48 + half * 8);
    acc[t] = __builtin_amdgcn_wmma_f32_16x16x32_bf16(false, a0.v, false, b0.v,
                                                     (short)0, acc[t], false, false);
    acc[t] = __builtin_amdgcn_wmma_f32_16x16x32_bf16(false, a1.v, false, b1.v,
                                                     (short)0, acc[t], false, false);
  }

  // ---- softmax over rows m = v + 8*half ----
  const float scale = 0.125f; // 1/sqrt(64)
  float rmax[8];
#pragma unroll
  for (int v = 0; v < 8; ++v) {
    float mx = -1e30f;
#pragma unroll
    for (int t = 0; t < 8; ++t) { acc[t][v] *= scale; mx = fmaxf(mx, acc[t][v]); }
    rmax[v] = mx;
  }
#pragma unroll
  for (int v = 0; v < 8; ++v)
    for (int d = 1; d < 16; d <<= 1)
      rmax[v] = fmaxf(rmax[v], __shfl_xor(rmax[v], d, 32));
  if (n == 0) {
#pragma unroll
    for (int v = 0; v < 8; ++v) red[wave][v + 8 * half] = rmax[v];
  }
  __syncthreads();
  if (tid < 16) {
    float mx = red[0][tid];
    for (int w = 1; w < 8; ++w) mx = fmaxf(mx, red[w][tid]);
    rowstat[tid] = mx;
  }
  __syncthreads();
  float gmax[8], rsum[8];
#pragma unroll
  for (int v = 0; v < 8; ++v) gmax[v] = rowstat[v + 8 * half];
#pragma unroll
  for (int v = 0; v < 8; ++v) {
    float s = 0.f;
#pragma unroll
    for (int t = 0; t < 8; ++t) {
      const float e = expf(acc[t][v] - gmax[v]);
      acc[t][v] = e;
      s += e;
    }
    rsum[v] = s;
  }
#pragma unroll
  for (int v = 0; v < 8; ++v)
    for (int d = 1; d < 16; d <<= 1)
      rsum[v] += __shfl_xor(rsum[v], d, 32);
  __syncthreads(); // protect red[] reuse
  if (n == 0) {
#pragma unroll
    for (int v = 0; v < 8; ++v) red[wave][v + 8 * half] = rsum[v];
  }
  __syncthreads();
  if (tid < 16) {
    float s = 0.f;
    for (int w = 0; w < 8; ++w) s += red[w][tid];
    rowstat[tid] = s;
  }
  __syncthreads();
  float inv[8];
#pragma unroll
  for (int v = 0; v < 8; ++v) inv[v] = 1.f / rowstat[v + 8 * half];
  for (int t = 0; t < 8; ++t) {
    const int ncol = (wave + t * 8) * 16 + n;
#pragma unroll
    for (int v = 0; v < 8; ++v)
      pb[v + 8 * half][ncol] = (bf16_t)(acc[t][v] * inv[v]);
  }
  __syncthreads();

  // ---- P * V : 4 output n-tiles (DK=64), waves 0..3; V is [B,H,DK,S] ----
  if (wave < 4) {
    v8f o = {0.f,0.f,0.f,0.f,0.f,0.f,0.f,0.f};
    const int ndk = wave * 16 + n;
    const bf16_t* vrow = vt + ((long long)(bb * Hc + hh) * DKc + ndk) * Sc;
    for (int kk = 0; kk < Sc; kk += 32) {
      Frag16 af, bf;
      ((uint4*)af.u)[0] = *(const uint4*)&pb[n][kk + 8 * half];
      ((uint4*)af.u)[1] = *(const uint4*)&pb[n][kk + 16 + 8 * half];
      ((uint4*)bf.u)[0] = *(const uint4*)(vrow + kk + half * 8);
      ((uint4*)bf.u)[1] = *(const uint4*)(vrow + kk + 16 + half * 8);
      o = __builtin_amdgcn_wmma_f32_16x16x32_bf16(false, af.v, false, bf.v,
                                                  (short)0, o, false, false);
    }
    const float wq = sigmf(gates[hh]);
#pragma unroll
    for (int v = 0; v < 8; ++v) {
      const float c = o[v];
      const float val = wq * sinf(c) + (1.f - wq) * c;
      ob[((long long)bb * Sc + q0 + v + 8 * half) * Dc + hh * DKc + wave * 16 + n] = val;
    }
  }
}

// ---------------- fused residual-add + LayerNorm (in-place, float4) ----------------
__global__ __launch_bounds__(256)
void k_ln(float* __restrict__ x, const float* __restrict__ add,
          const float* __restrict__ g, const float* __restrict__ bta) {
  __shared__ float sred[8], sred2[8];
  __shared__ float s_mu, s_rs;
  const int row = blockIdx.x, tid = threadIdx.x;
  float4* xrow = (float4*)(x + (long long)row * Dc);
  const float4 xv = xrow[tid];
  const float4 av = ((const float4*)(add + (long long)row * Dc))[tid];
  float4 v;
  v.x = xv.x + av.x; v.y = xv.y + av.y; v.z = xv.z + av.z; v.w = xv.w + av.w;
  float sum = v.x + v.y + v.z + v.w;
  float sumsq = v.x * v.x + v.y * v.y + v.z * v.z + v.w * v.w;
  for (int d = 1; d < 32; d <<= 1) {
    sum += __shfl_xor(sum, d, 32);
    sumsq += __shfl_xor(sumsq, d, 32);
  }
  if ((tid & 31) == 0) { sred[tid >> 5] = sum; sred2[tid >> 5] = sumsq; }
  __syncthreads();
  if (tid == 0) {
    float s = 0.f, q = 0.f;
    for (int w = 0; w < 8; ++w) { s += sred[w]; q += sred2[w]; }
    const float mu = s / (float)Dc;
    s_mu = mu;
    s_rs = rsqrtf(q / (float)Dc - mu * mu + 1e-5f);
  }
  __syncthreads();
  const float mu = s_mu, rs = s_rs;
  const float4 gv = ((const float4*)g)[tid];
  const float4 bv = ((const float4*)bta)[tid];
  float4 r;
  r.x = (v.x - mu) * rs * gv.x + bv.x;
  r.y = (v.y - mu) * rs * gv.y + bv.y;
  r.z = (v.z - mu) * rs * gv.z + bv.z;
  r.w = (v.w - mu) * rs * gv.w + bv.w;
  xrow[tid] = r;
}

// ---------------- mean-pool over S + classifier ----------------
__global__ void k_clf(const float* __restrict__ x, const float* __restrict__ W,
                      const float* __restrict__ cb, float* __restrict__ out) {
  __shared__ float sred[8];
  const int bb = blockIdx.x, tid = threadIdx.x;
  float acc = 0.f;
  for (int d = tid; d < Dc; d += 256) {
    const float w = W[d];
    float s = 0.f;
    for (int ss = 0; ss < Sc; ++ss) s += x[((long long)bb * Sc + ss) * Dc + d];
    acc += s * w;
  }
  for (int d = 1; d < 32; d <<= 1) acc += __shfl_xor(acc, d, 32);
  if ((tid & 31) == 0) sred[tid >> 5] = acc;
  __syncthreads();
  if (tid == 0) {
    float s = 0.f;
    for (int w = 0; w < 8; ++w) s += sred[w];
    out[bb] = s / (float)Sc + cb[0];
  }
}

// ---------------- workspace layout (bytes) ----------------
static constexpr size_t MB = 1ull << 20;
static constexpr size_t X_OFF   = 0;          // f32  BS*D   (16 MB)
static constexpr size_t XB_OFF  = 16 * MB;    // bf16 BS*D   ( 8 MB)
static constexpr size_t QB_OFF  = 24 * MB;    // bf16 BS*D
static constexpr size_t KB_OFF  = 32 * MB;    // bf16 BS*D
static constexpr size_t VT_OFF  = 40 * MB;    // bf16 [B,H,DK,S]
static constexpr size_t OB_OFF  = 48 * MB;    // f32  BS*D   (16 MB)
static constexpr size_t OBF_OFF = 64 * MB;    // bf16 BS*D
static constexpr size_t T_OFF   = 72 * MB;    // f32  BS*D   (16 MB)
static constexpr size_t HB_OFF  = 88 * MB;    // bf16 BS*F   (32 MB)
static constexpr size_t WBF_OFF = 120 * MB;   // bf16 weight scratch, transposed (8 MB)

extern "C" void kernel_launch(void* const* d_in, const int* in_sizes, int n_in,
                              void* d_out, int out_size, void* d_ws, size_t ws_size,
                              hipStream_t stream) {
  (void)in_sizes; (void)n_in; (void)out_size; (void)ws_size;
  const int*   tok    = (const int*)d_in[0];
  const float* emb    = (const float*)d_in[1];
  const float* pe_qw  = (const float*)d_in[2];
  const float* Wq     = (const float*)d_in[3];
  const float* Wk     = (const float*)d_in[4];
  const float* Wv     = (const float*)d_in[5];
  const float* Wo     = (const float*)d_in[6];
  const float* aqw    = (const float*)d_in[7];
  const float* ln1_g  = (const float*)d_in[8];
  const float* ln1_b  = (const float*)d_in[9];
  const float* ln2_g  = (const float*)d_in[10];
  const float* ln2_b  = (const float*)d_in[11];
  const float* W1     = (const float*)d_in[12];
  const float* b1     = (const float*)d_in[13];
  const float* W2     = (const float*)d_in[14];
  const float* b2     = (const float*)d_in[15];
  const float* ffn_qw = (const float*)d_in[16];
  const float* clfW   = (const float*)d_in[17];
  const float* clfb   = (const float*)d_in[18];

  char* ws = (char*)d_ws;
  float*  X   = (float*)(ws + X_OFF);
  bf16_t* XB  = (bf16_t*)(ws + XB_OFF);
  bf16_t* QB  = (bf16_t*)(ws + QB_OFF);
  bf16_t* KB  = (bf16_t*)(ws + KB_OFF);
  bf16_t* VT  = (bf16_t*)(ws + VT_OFF);
  float*  OB  = (float*)(ws + OB_OFF);
  bf16_t* OBF = (bf16_t*)(ws + OBF_OFF);
  float*  T   = (float*)(ws + T_OFF);
  bf16_t* HB  = (bf16_t*)(ws + HB_OFF);
  bf16_t* WBF = (bf16_t*)(ws + WBF_OFF);

  const int nXD = BS * Dc;      // 4M
  const int nDD = Dc * Dc;      // 1M
  const dim3 blk(256);

  k_embed<<<BS, blk, 0, stream>>>(tok, emb, pe_qw, X);

  for (int l = 0; l < Lc; ++l) {
    const float* Wq_l = Wq + (long long)l * nDD;
    const float* Wk_l = Wk + (long long)l * nDD;
    const float* Wv_l = Wv + (long long)l * nDD;
    const float* Wo_l = Wo + (long long)l * nDD;

    // x -> bf16
    k_cvt4<<<(nXD / 4 + 255) / 256, blk, 0, stream>>>(X, XB, nXD / 4);

    // Q, K, V projections (weights transposed to [N,K] bf16 first)
    k_cvtT<<<dim3(Dc / 32, Dc / 32), blk, 0, stream>>>(Wq_l, WBF, Dc, Dc);
    k_gemm<<<dim3(Dc / 64, BS / 128), blk, 0, stream>>>(XB, WBF, QB, nullptr, nullptr,
                                                        Dc, Dc, Dc, 1);
    k_cvtT<<<dim3(Dc / 32, Dc / 32), blk, 0, stream>>>(Wk_l, WBF, Dc, Dc);
    k_gemm<<<dim3(Dc / 64, BS / 128), blk, 0, stream>>>(XB, WBF, KB, nullptr, nullptr,
                                                        Dc, Dc, Dc, 1);
    k_cvtT<<<dim3(Dc / 32, Dc / 32), blk, 0, stream>>>(Wv_l, WBF, Dc, Dc);
    k_gemm<<<dim3(Dc / 64, BS / 128), blk, 0, stream>>>(XB, WBF, VT, nullptr, nullptr,
                                                        Dc, Dc, Dc, 3); // V transposed

    // attention with fused per-head sin gate
    k_attn<<<dim3(Sc / 16, Hc, Bc), blk, 0, stream>>>(QB, KB, VT, OB, aqw + l * Hc);

    // output projection
    k_cvt4<<<(nXD / 4 + 255) / 256, blk, 0, stream>>>(OB, OBF, nXD / 4);
    k_cvtT<<<dim3(Dc / 32, Dc / 32), blk, 0, stream>>>(Wo_l, WBF, Dc, Dc);
    k_gemm<<<dim3(Dc / 64, BS / 128), blk, 0, stream>>>(OBF, WBF, T, nullptr, nullptr,
                                                        Dc, Dc, Dc, 0);
    // x = LN1(x + attn_out)
    k_ln<<<BS, blk, 0, stream>>>(X, T, ln1_g + l * Dc, ln1_b + l * Dc);

    // FFN: h = hybrid_act(x @ W1 + b1)  (fused epilogue, bf16 out)
    k_cvt4<<<(nXD / 4 + 255) / 256, blk, 0, stream>>>(X, XB, nXD / 4);
    k_cvtT<<<dim3(Dc / 32, Fc / 32), blk, 0, stream>>>(W1 + (long long)l * Dc * Fc,
                                                       WBF, Dc, Fc);
    k_gemm<<<dim3(Fc / 64, BS / 128), blk, 0, stream>>>(XB, WBF, HB, b1 + l * Fc,
                                                        ffn_qw + l, Dc, Dc, Fc, 2);
    // ffn_out = h @ W2 + b2
    k_cvtT<<<dim3(Fc / 32, Dc / 32), blk, 0, stream>>>(W2 + (long long)l * Fc * Dc,
                                                       WBF, Fc, Dc);
    k_gemm<<<dim3(Dc / 64, BS / 128), blk, 0, stream>>>(HB, WBF, T, b2 + l * Dc,
                                                        nullptr, Fc, Fc, Dc, 0);
    // x = LN2(x + ffn_out)
    k_ln<<<BS, blk, 0, stream>>>(X, T, ln2_g + l * Dc, ln2_b + l * Dc);
  }

  k_clf<<<Bc, blk, 0, stream>>>(X, clfW, clfb, (float*)d_out);
}
